// AutoEncoder_9311489098138
// MI455X (gfx1250) — matrix operand order repaired and checked by reading, not compiled, not run on hardware
//
#include <hip/hip_runtime.h>
#include <hip/hip_bf16.h>

typedef _Float16 half_t;
typedef __attribute__((ext_vector_type(16))) _Float16 v16h;
typedef __attribute__((ext_vector_type(8)))  _Float16 v8h;
typedef __attribute__((ext_vector_type(4)))  _Float16 v4h;
typedef __attribute__((ext_vector_type(8)))  float    v8f;
typedef __attribute__((ext_vector_type(4)))  float    v4f;
typedef __attribute__((ext_vector_type(4)))  unsigned int v4u;
typedef __attribute__((ext_vector_type(8)))  int      v8i;
typedef __attribute__((ext_vector_type(4)))  int      v4i;

#define B_SETS   65536
#define NPER     8
#define DIM      64
#define HID      128
#define MAXN     16
#define MID_PE   72
#define MID_PSI  96
#define MID_PHI  136
#define MID_SP   72
#define MID_DEC  96

// padded dims for WMMA granularity (K -> mult of 32, N -> mult of 16)
#define PHI_N1P  144
#define PHI_K2P  160
#define SP_N1P   80
#define SP_K2P   96

// packed-fragment weight sizes (halves) == Kp*Np
#define SZ_PSI1  (64*96)
#define SZ_PSI2  (96*128)
#define SZ_PHI1  (128*PHI_N1P)
#define SZ_PHI2  (PHI_K2P*128)
#define SZ_SP1   (128*SP_N1P)
#define SZ_SP2   (SP_K2P*16)
#define SZ_DEC1  (128*96)
#define SZ_DEC2  (96*64)

// ---------------- workspace layout (byte offsets) ----------------
#define OFF_W_PSI1 ((size_t)0)
#define OFF_W_PSI2 (OFF_W_PSI1 + (size_t)SZ_PSI1*2)
#define OFF_W_PHI1 (OFF_W_PSI2 + (size_t)SZ_PSI2*2)
#define OFF_W_PHI2 (OFF_W_PHI1 + (size_t)SZ_PHI1*2)
#define OFF_W_SP1  (OFF_W_PHI2 + (size_t)SZ_PHI2*2)
#define OFF_W_SP2  (OFF_W_SP1  + (size_t)SZ_SP1*2)
#define OFF_W_DEC1 (OFF_W_SP2  + (size_t)SZ_SP2*2)
#define OFF_W_DEC2 (OFF_W_DEC1 + (size_t)SZ_DEC1*2)
#define OFF_PESEG  (OFF_W_DEC2 + (size_t)SZ_DEC2*2)
#define OFF_PEALL  (OFF_PESEG  + (size_t)NPER*HID*2)
#define OFF_PHIB1F (OFF_PEALL  + (size_t)MAXN*HID*2)
#define OFF_Y2     (((OFF_PHIB1F + (size_t)MID_PHI*4) + 255) & ~(size_t)255)
#define OFF_Z      (OFF_Y2 + (size_t)B_SETS*HID*2)
#define OFF_NDEC   (OFF_Z  + (size_t)B_SETS*HID*2)
#define WS_TOTAL   (OFF_NDEC + (size_t)B_SETS*4)

// ---------------- TDM: 1-row tensor DMA  global -> LDS ----------------------
// D# layout per CDNA5 ISA ch.8: group0 = {flags/count, lds_addr, gaddr_lo,
// gaddr_hi|type=2}; group1 packs data_size=8B, tensor_dim0 = tile_dim0 =
// nbytes/8, tensor_dim1 = tile_dim1 = 1. Groups 2/3 unused (zero).
// This toolchain exposes the 6-arg builtin (extra int32x8 group, then cpol).
__device__ __forceinline__ void tdm_load(void* lds_dst, const void* gsrc,
                                         unsigned nbytes) {
    unsigned lds_addr = (unsigned)(uintptr_t)lds_dst;   // low 32b = LDS offset
    unsigned long long ga = (unsigned long long)(uintptr_t)gsrc;
    unsigned n8 = nbytes >> 3;                          // 8-byte elements
    v4u g0;
    g0[0] = 1u;                                          // count=1 (user D#)
    g0[1] = lds_addr;
    g0[2] = (unsigned)(ga & 0xFFFFFFFFu);
    g0[3] = (unsigned)((ga >> 32) & 0x01FFFFFFu) | (2u << 30);  // type=2
    v8i g1;
    g1[0] = (int)(3u << 16);                             // data_size = 8B
    g1[1] = (int)((n8 & 0xFFFFu) << 16);                 // tensor_dim0[15:0]
    g1[2] = (int)(((n8 >> 16) & 0xFFFFu) | (1u << 16));  // dim0 hi | tensor_dim1=1
    g1[3] = (int)((n8 & 0xFFFFu) << 16);                 // tile_dim0 = n8
    g1[4] = 1;                                           // tile_dim1 = 1
    g1[5] = (int)n8;                                     // tensor_dim0_stride
    g1[6] = 0;
    g1[7] = 0;
    v4i g2 = {0,0,0,0};
    v4i g3 = {0,0,0,0};
    v8i g4 = {0,0,0,0,0,0,0,0};
    __builtin_amdgcn_tensor_load_to_lds(g0, g1, g2, g3, g4, 0);
}

__device__ __forceinline__ void tdm_wait_all() {
    __builtin_amdgcn_s_wait_tensorcnt(0);   // per-wave; issuer wave's wait matters
    __syncthreads();
}

// ---------------- WMMA fragment helpers (wave32 layouts per CDNA5 ISA) ------
__device__ __forceinline__ v16h load_A_frag(const half_t* p0, int lda) {
    const int l = threadIdx.x & 31;
    const half_t* p = p0 + (l & 15) * lda + ((l & 16) ? 8 : 0);
    v8h lo = *(const v8h*)p;          // K = kb .. kb+7
    v8h hi = *(const v8h*)(p + 16);   // K = 16+kb .. 16+kb+7
    return __builtin_shufflevector(lo, hi, 0,1,2,3,4,5,6,7,8,9,10,11,12,13,14,15);
}

__device__ __forceinline__ v16h load_B_frag_pk(const half_t* tile) {
    const int l = threadIdx.x & 31;
    const half_t* p = tile + l * 16;
    v8h lo = *(const v8h*)p;
    v8h hi = *(const v8h*)(p + 8);
    return __builtin_shufflevector(lo, hi, 0,1,2,3,4,5,6,7,8,9,10,11,12,13,14,15);
}

__device__ __forceinline__ v8f wmma_tile_pk(const half_t* A, int lda,
                                            const half_t* Wpk, int Ntiles,
                                            int row0, int nt, int K) {
    v8f c = {};
    int kt = 0;
    for (int k = 0; k < K; k += 32, ++kt) {
        v16h a = load_A_frag(A + row0 * lda + k, lda);
        v16h b = load_B_frag_pk(Wpk + ((size_t)(kt * Ntiles + nt) << 9));
        c = __builtin_amdgcn_wmma_f32_16x16x32_f16(false, a, false, b,
                                                   (short)0, c, false, false);
    }
    return c;
}

__device__ __forceinline__ void copy_f(float* dst, const float* src, int n) {
    for (int i = threadIdx.x; i < n; i += (int)blockDim.x) dst[i] = src[i];
}

// ---------------- K0: pack weights into fragment-major f16 + PE tables -------
__device__ void pack_B(half_t* dst, const float* src, int K, int N, int Kp, int Np,
                       int t, int NT) {
    const int Ntiles = Np >> 4;
    const int total = Kp * Np;
    for (int idx = t; idx < total; idx += NT) {
        int i    = idx & 15;
        int lane = (idx >> 4) & 31;
        int tile = idx >> 9;
        int nt = tile % Ntiles, kt = tile / Ntiles;
        int k = kt * 32 + ((lane & 16) ? 8 : 0) + (i < 8 ? i : i + 8);
        int n = nt * 16 + (lane & 15);
        float v = (k < K && n < N) ? src[k * N + n] : 0.f;
        dst[idx] = (half_t)v;
    }
}

__global__ __launch_bounds__(256) void prep_kernel(
    const float* epe_w1, const float* epe_b1, const float* epe_w2, const float* epe_b2,
    const float* psi_w1, const float* psi_w2,
    const float* phi_w1, const float* phi_b1, const float* phi_w2,
    const float* sp_w1, const float* sp_w2,
    const float* dpe_w1, const float* dpe_b1, const float* dpe_w2, const float* dpe_b2,
    const float* dec_w1, const float* dec_w2,
    char* ws)
{
    const int t  = blockIdx.x * blockDim.x + threadIdx.x;
    const int NT = gridDim.x * blockDim.x;
    half_t* wPSI1 = (half_t*)(ws + OFF_W_PSI1);
    half_t* wPSI2 = (half_t*)(ws + OFF_W_PSI2);
    half_t* wPHI1 = (half_t*)(ws + OFF_W_PHI1);
    half_t* wPHI2 = (half_t*)(ws + OFF_W_PHI2);
    half_t* wSP1  = (half_t*)(ws + OFF_W_SP1);
    half_t* wSP2  = (half_t*)(ws + OFF_W_SP2);
    half_t* wDEC1 = (half_t*)(ws + OFF_W_DEC1);
    half_t* wDEC2 = (half_t*)(ws + OFF_W_DEC2);
    half_t* peSEG = (half_t*)(ws + OFF_PESEG);
    half_t* peALL = (half_t*)(ws + OFF_PEALL);
    float*  phiB1 = (float*)(ws + OFF_PHIB1F);

    pack_B(wPSI1, psi_w1, DIM,     MID_PSI, DIM,     MID_PSI, t, NT);
    pack_B(wPSI2, psi_w2, MID_PSI, HID,     MID_PSI, HID,     t, NT);
    pack_B(wPHI1, phi_w1, HID,     MID_PHI, HID,     PHI_N1P, t, NT);  // first 128 K-rows
    pack_B(wPHI2, phi_w2, MID_PHI, HID,     PHI_K2P, HID,     t, NT);
    pack_B(wSP1,  sp_w1,  HID,     MID_SP,  HID,     SP_N1P,  t, NT);
    pack_B(wSP2,  sp_w2,  MID_SP,  MAXN,    SP_K2P,  MAXN,    t, NT);
    pack_B(wDEC1, dec_w1, HID,     MID_DEC, HID,     MID_DEC, t, NT);
    pack_B(wDEC2, dec_w2, MID_DEC, DIM,     MID_DEC, DIM,     t, NT);

    // fold constant pos_n (one-hot at NPER=8) into phi layer-1 bias
    for (int j = t; j < MID_PHI; j += NT)
        phiB1[j] = phi_b1[j] + phi_w1[(HID + NPER)*MID_PHI + j];

    // pe_seg[i<8][j] = relu(epe_w1[i,:]+b1) @ epe_w2 + b2   (one-hot input)
    for (int o = t; o < NPER*HID; o += NT) {
        int i = o / HID, j = o % HID;
        float acc = epe_b2[j];
        for (int h = 0; h < MID_PE; ++h) {
            float hv = fmaxf(epe_w1[i*MID_PE + h] + epe_b1[h], 0.f);
            acc += hv * epe_w2[h*HID + j];
        }
        peSEG[o] = (half_t)acc;
    }
    // pe_all[i<16][j] with decoder pos encoder
    for (int o = t; o < MAXN*HID; o += NT) {
        int i = o / HID, j = o % HID;
        float acc = dpe_b2[j];
        for (int h = 0; h < MID_PE; ++h) {
            float hv = fmaxf(dpe_w1[i*MID_PE + h] + dpe_b1[h], 0.f);
            acc += hv * dpe_w2[h*HID + j];
        }
        peALL[o] = (half_t)acc;
    }
}

// ---------------- K1: rank + per-set sort + psi MLP + PE scale + segment sum --
// 16 sets (128 rows) per block, 256 threads = 8 waves.
__global__ __launch_bounds__(256) void enc_psi_kernel(
    const float* __restrict__ x,
    const float* __restrict__ rank_w, const float* __restrict__ rank_b,
    const float* __restrict__ psi_b1, const float* __restrict__ psi_b2,
    const half_t* __restrict__ wPSI1, const half_t* __restrict__ wPSI2,
    const half_t* __restrict__ peSEG,
    half_t* __restrict__ Y2)
{
    __shared__ __attribute__((aligned(16))) half_t sW1[SZ_PSI1];
    __shared__ __attribute__((aligned(16))) half_t sW2[SZ_PSI2];
    __shared__ __attribute__((aligned(16))) half_t sPE[NPER*HID];
    __shared__ float  sB1[MID_PSI];
    __shared__ float  sB2[HID];
    __shared__ __attribute__((aligned(16))) half_t sX[128*DIM];
    __shared__ __attribute__((aligned(16))) half_t sMid[128*MID_PSI];
    __shared__ __attribute__((aligned(16))) half_t sY1[128*HID];
    __shared__ float  sMag[128];
    __shared__ short  sOrd[128];

    const int t = threadIdx.x, lane = t & 31, wave = t >> 5;
    const int rowbase = blockIdx.x * 128;

    if (wave == 0) {                       // DMA weight/PE staging via TDM
        tdm_load(sW1, wPSI1, SZ_PSI1*2);
        tdm_load(sW2, wPSI2, SZ_PSI2*2);
        tdm_load(sPE, peSEG, NPER*HID*2);
    }
    copy_f(sB1, psi_b1, MID_PSI);
    copy_f(sB2, psi_b2, HID);

    // learned rank per row (+ prefetch next block's x tile into L2)
    if (t < 128) {
        const float* xr = x + (size_t)(rowbase + t) * DIM;
        if (blockIdx.x + 1 < gridDim.x)
            __builtin_prefetch(xr + 128 * DIM, 0, 1);
        const v4f* xv = (const v4f*)xr;
        const v4f* rw = (const v4f*)rank_w;
        float acc = rank_b[0];
        for (int c = 0; c < DIM/4; ++c) {
            v4f a = xv[c], b = rw[c];
            acc += a[0]*b[0] + a[1]*b[1] + a[2]*b[2] + a[3]*b[3];
        }
        sMag[t] = acc;
    }
    __syncthreads();

    // stable ascending argsort of 8 per set (selection, ties keep lower index)
    if (t < 16) {
        const int base = t * NPER;
        unsigned used = 0;
        for (int i = 0; i < NPER; ++i) {
            int best = -1; float bv = 0.f;
            for (int j = 0; j < NPER; ++j) {
                if (used & (1u << j)) continue;
                float v = sMag[base + j];
                if (best < 0 || v < bv) { best = j; bv = v; }
            }
            used |= 1u << best;
            sOrd[base + i] = (short)(base + best);
        }
    }
    __syncthreads();

    // gather sorted rows -> f16 (16B global loads, 8B LDS stores)
    for (int i = t; i < 128*16; i += 256) {
        int r = i >> 4, c0 = (i & 15) << 2;
        v4f xv = *(const v4f*)&x[(size_t)(rowbase + sOrd[r]) * DIM + c0];
        v4h o;
#pragma unroll
        for (int j = 0; j < 4; ++j) o[j] = (half_t)xv[j];
        *(v4h*)&sX[r*DIM + c0] = o;
    }
    tdm_wait_all();

    // layer 1: [128x64] @ [64x96], ReLU
    for (int tile = wave; tile < 8*6; tile += 8) {
        int mi = tile / 6, ni = tile % 6;
        v8f c = wmma_tile_pk(sX, DIM, sW1, 6, mi*16, ni, DIM);
        int rb = mi*16 + ((lane >> 4) << 3), cc = ni*16 + (lane & 15);
#pragma unroll
        for (int r = 0; r < 8; ++r)
            sMid[(rb + r)*MID_PSI + cc] = (half_t)fmaxf(c[r] + sB1[cc], 0.f);
    }
    __syncthreads();

    // layer 2: [128x96] @ [96x128], then * pe_seg[row%8]
    for (int tile = wave; tile < 8*8; tile += 8) {
        int mi = tile >> 3, ni = tile & 7;
        v8f c = wmma_tile_pk(sMid, MID_PSI, sW2, 8, mi*16, ni, MID_PSI);
        int rb = mi*16 + ((lane >> 4) << 3), cc = ni*16 + (lane & 15);
#pragma unroll
        for (int r = 0; r < 8; ++r) {
            int rr = rb + r;
            float v = (c[r] + sB2[cc]) * (float)sPE[(rr & 7)*HID + cc];
            sY1[rr*HID + cc] = (half_t)v;
        }
    }
    __syncthreads();

    // segment sum over 8 rows -> y2; one 8-col chunk per thread, b128 store
    {
        int s = t >> 4, c0 = (t & 15) << 3;
        float acc[8] = {0,0,0,0,0,0,0,0};
#pragma unroll
        for (int r = 0; r < NPER; ++r) {
            v8h y = *(const v8h*)&sY1[(s*NPER + r)*HID + c0];
#pragma unroll
            for (int j = 0; j < 8; ++j) acc[j] += (float)y[j];
        }
        v8h out;
#pragma unroll
        for (int j = 0; j < 8; ++j) out[j] = (half_t)acc[j];
        *(v8h*)&Y2[(size_t)(blockIdx.x*16 + s)*HID + c0] = out;
    }
}

// ---------------- K2: phi MLP (pos folded) + size_pred MLP + argmax ----------
// 32 sets per block.
__global__ __launch_bounds__(256) void enc_phi_sp_kernel(
    const half_t* __restrict__ Y2,
    const half_t* __restrict__ wPHI1, const half_t* __restrict__ wPHI2,
    const half_t* __restrict__ wSP1,  const half_t* __restrict__ wSP2,
    const float* __restrict__ phiB1f, const float* __restrict__ phi_b2,
    const float* __restrict__ sp_b1,  const float* __restrict__ sp_b2,
    half_t* __restrict__ Z, int* __restrict__ ndec_ws, int* __restrict__ ndec_out)
{
    __shared__ __attribute__((aligned(16))) half_t sW1[SZ_PHI1];
    __shared__ __attribute__((aligned(16))) half_t sW2[SZ_PHI2];
    __shared__ __attribute__((aligned(16))) half_t sSP1[SZ_SP1];
    __shared__ __attribute__((aligned(16))) half_t sSP2[SZ_SP2];
    __shared__ float  sB1[MID_PHI], sB2[HID], sSB1[MID_SP], sSB2[MAXN];
    __shared__ __attribute__((aligned(16))) half_t sY2[32*HID];
    __shared__ __attribute__((aligned(16))) half_t sMid[32*PHI_K2P];
    __shared__ __attribute__((aligned(16))) half_t sZ[32*HID];
    __shared__ __attribute__((aligned(16))) half_t sSM[32*SP_K2P];
    __shared__ float  sNP[32*MAXN];

    const int t = threadIdx.x, lane = t & 31, wave = t >> 5;
    const int setbase = blockIdx.x * 32;

    if (wave == 0) {
        tdm_load(sW1, wPHI1, SZ_PHI1*2);
        tdm_load(sW2, wPHI2, SZ_PHI2*2);
        tdm_load(sSP1, wSP1, SZ_SP1*2);
        tdm_load(sSP2, wSP2, SZ_SP2*2);
        tdm_load(sY2, Y2 + (size_t)setbase*HID, 32*HID*2);
    }
    copy_f(sB1, phiB1f, MID_PHI);
    copy_f(sB2, phi_b2, HID);
    copy_f(sSB1, sp_b1, MID_SP);
    copy_f(sSB2, sp_b2, MAXN);
    // zero pad tails the GEMM epilogues don't cover
    for (int i = t; i < 32*(PHI_K2P - PHI_N1P)/8; i += 256) {
        int r = i / 2, c = (i & 1) * 8;
        *(v8h*)&sMid[r*PHI_K2P + PHI_N1P + c] = (v8h){};
    }
    for (int i = t; i < 32*(SP_K2P - SP_N1P)/8; i += 256) {
        int r = i / 2, c = (i & 1) * 8;
        *(v8h*)&sSM[r*SP_K2P + SP_N1P + c] = (v8h){};
    }
    tdm_wait_all();

    // phi layer 1: [32x128] @ [128x144], ReLU (cols >= 136 forced 0)
    for (int tile = wave; tile < 2*9; tile += 8) {
        int mi = tile / 9, ni = tile % 9;
        v8f c = wmma_tile_pk(sY2, HID, sW1, 9, mi*16, ni, HID);
        int rb = mi*16 + ((lane >> 4) << 3), cc = ni*16 + (lane & 15);
#pragma unroll
        for (int r = 0; r < 8; ++r) {
            float v = (cc < MID_PHI) ? fmaxf(c[r] + sB1[cc], 0.f) : 0.f;
            sMid[(rb + r)*PHI_K2P + cc] = (half_t)v;
        }
    }
    __syncthreads();

    // phi layer 2: [32x160] @ [160x128] -> z
    for (int tile = wave; tile < 2*8; tile += 8) {
        int mi = tile >> 3, ni = tile & 7;
        v8f c = wmma_tile_pk(sMid, PHI_K2P, sW2, 8, mi*16, ni, PHI_K2P);
        int rb = mi*16 + ((lane >> 4) << 3), cc = ni*16 + (lane & 15);
#pragma unroll
        for (int r = 0; r < 8; ++r) {
            int rr = rb + r;
            half_t v = (half_t)(c[r] + sB2[cc]);
            sZ[rr*HID + cc] = v;
            Z[(size_t)(setbase + rr)*HID + cc] = v;
        }
    }
    __syncthreads();

    // sp layer 1: [32x128] @ [128x80], ReLU (cols >= 72 forced 0)
    for (int tile = wave; tile < 2*5; tile += 8) {
        int mi = tile / 5, ni = tile % 5;
        v8f c = wmma_tile_pk(sZ, HID, sSP1, 5, mi*16, ni, HID);
        int rb = mi*16 + ((lane >> 4) << 3), cc = ni*16 + (lane & 15);
#pragma unroll
        for (int r = 0; r < 8; ++r) {
            float v = (cc < MID_SP) ? fmaxf(c[r] + sSB1[cc], 0.f) : 0.f;
            sSM[(rb + r)*SP_K2P + cc] = (half_t)v;
        }
    }
    __syncthreads();

    // sp layer 2: [32x96] @ [96x16] -> logits
    for (int tile = wave; tile < 2; tile += 8) {
        v8f c = wmma_tile_pk(sSM, SP_K2P, sSP2, 1, tile*16, 0, SP_K2P);
        int rb = tile*16 + ((lane >> 4) << 3), cc = lane & 15;
#pragma unroll
        for (int r = 0; r < 8; ++r)
            sNP[(rb + r)*MAXN + cc] = c[r] + sSB2[cc];
    }
    __syncthreads();

    // argmax (first max wins, matching jnp.argmax)
    if (t < 32) {
        int am = 0; float bv = sNP[t*MAXN];
        for (int j = 1; j < MAXN; ++j) {
            float v = sNP[t*MAXN + j];
            if (v > bv) { bv = v; am = j; }
        }
        ndec_ws[setbase + t]  = am;
        ndec_out[setbase + t] = am;
    }
}

// ---------------- K3: decoder MLP with pe_all broadcast + mask ----------------
// 8 sets per block -> 128 zp rows.
__global__ __launch_bounds__(256) void dec_kernel(
    const half_t* __restrict__ Z, const int* __restrict__ ndec,
    const half_t* __restrict__ peALL,
    const half_t* __restrict__ wDEC1, const half_t* __restrict__ wDEC2,
    const float* __restrict__ dec_b1, const float* __restrict__ dec_b2,
    float* __restrict__ xr)
{
    __shared__ __attribute__((aligned(16))) half_t sW1[SZ_DEC1];
    __shared__ __attribute__((aligned(16))) half_t sW2[SZ_DEC2];
    __shared__ __attribute__((aligned(16))) half_t sPE[MAXN*HID];
    __shared__ __attribute__((aligned(16))) half_t sZ[8*HID];
    __shared__ __attribute__((aligned(16))) half_t sZP[128*HID];
    __shared__ __attribute__((aligned(16))) half_t sMid[128*MID_DEC];
    __shared__ float  sB1[MID_DEC], sB2[DIM];
    __shared__ int    sN[8];

    const int t = threadIdx.x, lane = t & 31, wave = t >> 5;
    const int setbase = blockIdx.x * 8;

    if (wave == 0) {
        tdm_load(sW1, wDEC1, SZ_DEC1*2);
        tdm_load(sW2, wDEC2, SZ_DEC2*2);
        tdm_load(sPE, peALL, MAXN*HID*2);
        tdm_load(sZ, Z + (size_t)setbase*HID, 8*HID*2);
    }
    copy_f(sB1, dec_b1, MID_DEC);
    copy_f(sB2, dec_b2, DIM);
    if (t < 8) sN[t] = ndec[setbase + t];
    tdm_wait_all();

    // zp[s,j,:] = z[s] * pe_all[j]  (vectorized, 8 halves per op)
    for (int i = t; i < 128*16; i += 256) {
        int r = i >> 4, c0 = (i & 15) << 3;
        v8h zv = *(const v8h*)&sZ[(r >> 4)*HID + c0];
        v8h pv = *(const v8h*)&sPE[(r & 15)*HID + c0];
        *(v8h*)&sZP[r*HID + c0] = zv * pv;
    }
    __syncthreads();

    // layer 1: [128x128] @ [128x96], ReLU
    for (int tile = wave; tile < 8*6; tile += 8) {
        int mi = tile / 6, ni = tile % 6;
        v8f c = wmma_tile_pk(sZP, HID, sW1, 6, mi*16, ni, HID);
        int rb = mi*16 + ((lane >> 4) << 3), cc = ni*16 + (lane & 15);
#pragma unroll
        for (int r = 0; r < 8; ++r)
            sMid[(rb + r)*MID_DEC + cc] = (half_t)fmaxf(c[r] + sB1[cc], 0.f);
    }
    __syncthreads();

    // layer 2: [128x96] @ [96x64] + size mask, write xr
    for (int tile = wave; tile < 8*4; tile += 8) {
        int mi = tile >> 2, ni = tile & 3;
        v8f c = wmma_tile_pk(sMid, MID_DEC, sW2, 4, mi*16, ni, MID_DEC);
        int rb = mi*16 + ((lane >> 4) << 3), cc = ni*16 + (lane & 15);
#pragma unroll
        for (int r = 0; r < 8; ++r) {
            int rr = rb + r;
            int s = rr >> 4, j = rr & 15;
            float v = c[r] + sB2[cc];
            if (j >= sN[s]) v = 0.f;
            xr[(size_t)(blockIdx.x*128 + rr)*DIM + cc] = v;
        }
    }
}

// ---------------- host launcher ----------------
extern "C" void kernel_launch(void* const* d_in, const int* in_sizes, int n_in,
                              void* d_out, int out_size, void* d_ws, size_t ws_size,
                              hipStream_t stream) {
    (void)in_sizes; (void)n_in; (void)out_size;
    if (ws_size < WS_TOTAL) return;

    const float* x      = (const float*)d_in[0];
    // d_in[1] = batch (segments are equal-size; unused)
    const float* rank_w = (const float*)d_in[2];
    const float* rank_b = (const float*)d_in[3];
    const float* epe_w1 = (const float*)d_in[4];
    const float* epe_b1 = (const float*)d_in[5];
    const float* epe_w2 = (const float*)d_in[6];
    const float* epe_b2 = (const float*)d_in[7];
    const float* psi_w1 = (const float*)d_in[8];
    const float* psi_b1 = (const float*)d_in[9];
    const float* psi_w2 = (const float*)d_in[10];
    const float* psi_b2 = (const float*)d_in[11];
    const float* phi_w1 = (const float*)d_in[12];
    const float* phi_b1 = (const float*)d_in[13];
    const float* phi_w2 = (const float*)d_in[14];
    const float* phi_b2 = (const float*)d_in[15];
    const float* sp_w1  = (const float*)d_in[16];
    const float* sp_b1  = (const float*)d_in[17];
    const float* sp_w2  = (const float*)d_in[18];
    const float* sp_b2  = (const float*)d_in[19];
    const float* dpe_w1 = (const float*)d_in[20];
    const float* dpe_b1 = (const float*)d_in[21];
    const float* dpe_w2 = (const float*)d_in[22];
    const float* dpe_b2 = (const float*)d_in[23];
    const float* dec_w1 = (const float*)d_in[24];
    const float* dec_b1 = (const float*)d_in[25];
    const float* dec_w2 = (const float*)d_in[26];
    const float* dec_b2 = (const float*)d_in[27];

    char* ws = (char*)d_ws;
    half_t* wPSI1 = (half_t*)(ws + OFF_W_PSI1);
    half_t* wPSI2 = (half_t*)(ws + OFF_W_PSI2);
    half_t* wPHI1 = (half_t*)(ws + OFF_W_PHI1);
    half_t* wPHI2 = (half_t*)(ws + OFF_W_PHI2);
    half_t* wSP1  = (half_t*)(ws + OFF_W_SP1);
    half_t* wSP2  = (half_t*)(ws + OFF_W_SP2);
    half_t* wDEC1 = (half_t*)(ws + OFF_W_DEC1);
    half_t* wDEC2 = (half_t*)(ws + OFF_W_DEC2);
    half_t* peSEG = (half_t*)(ws + OFF_PESEG);
    half_t* peALL = (half_t*)(ws + OFF_PEALL);
    float*  phiB1 = (float*)(ws + OFF_PHIB1F);
    half_t* Y2    = (half_t*)(ws + OFF_Y2);
    half_t* Z     = (half_t*)(ws + OFF_Z);
    int*    NDEC  = (int*)(ws + OFF_NDEC);

    float* xr = (float*)d_out;
    int* ndec_out = (int*)((float*)d_out + (size_t)B_SETS*MAXN*DIM);

    prep_kernel<<<16, 256, 0, stream>>>(
        epe_w1, epe_b1, epe_w2, epe_b2, psi_w1, psi_w2,
        phi_w1, phi_b1, phi_w2, sp_w1, sp_w2,
        dpe_w1, dpe_b1, dpe_w2, dpe_b2, dec_w1, dec_w2, ws);

    enc_psi_kernel<<<B_SETS/16, 256, 0, stream>>>(
        x, rank_w, rank_b, psi_b1, psi_b2, wPSI1, wPSI2, peSEG, Y2);

    enc_phi_sp_kernel<<<B_SETS/32, 256, 0, stream>>>(
        Y2, wPHI1, wPHI2, wSP1, wSP2, phiB1, phi_b2, sp_b1, sp_b2,
        Z, NDEC, ndec_out);

    dec_kernel<<<B_SETS/8, 256, 0, stream>>>(
        Z, NDEC, peALL, wDEC1, wDEC2, dec_b1, dec_b2, xr);
}
